// Attention_86474871538374
// MI455X (gfx1250) — matrix operand order repaired and checked by reading, not compiled, not run on hardware
//
#include <hip/hip_runtime.h>

// ---------------------------------------------------------------------------
// Problem constants (from reference): B=2, S=2048, E=2048, H=16, KV=4, D=128
// ---------------------------------------------------------------------------
#define BATCH 2
#define SEQ   2048
#define EMB   2048
#define NH    16
#define NKV   4
#define HD    128
#define WIN   512
#define QKVN  ((NH + 2 * NKV) * HD)   // 3072
#define SCALE 0.08838834764831845f    // 1/sqrt(128)

typedef __attribute__((ext_vector_type(16))) __bf16 v16bf;
typedef __attribute__((ext_vector_type(8)))  float  v8f;

// ---------------------------------------------------------------------------
// WMMA helpers — v_wmma_f32_16x16x32_bf16, D = A(16x32) * B(32x16) + C
// Layouts per CDNA5 ISA 7.12.2:
//   A: lane = m(0..15) + 16*half; elem e -> k = 8*half + e (e<8), 16+8*half+e-8
//   B: lane = n(0..15) + 16*half; elem e -> k = 8*half + e (e<8), 16+8*half+e-8
//   C/D: lane = n + 16*half; vgpr r -> m = r + 8*half
// ---------------------------------------------------------------------------
__device__ __forceinline__ v8f wmma_bf16(v16bf a, v16bf b, v8f c) {
  return __builtin_amdgcn_wmma_f32_16x16x32_bf16(
      false, a, false, b, (short)0, c, false, false);
}

// A-fragment: base points at element (m=0, k=0) of a 16x32 tile, row stride ldk
__device__ __forceinline__ v16bf frag_a(const __bf16* base, int ldk, int lane) {
  const int m = lane & 15, half = lane >> 4;
  const __bf16* p = base + (size_t)m * ldk + 8 * half;
  v16bf a;
#pragma unroll
  for (int e = 0; e < 8; ++e) a[e] = p[e];
#pragma unroll
  for (int e = 0; e < 8; ++e) a[8 + e] = p[16 + e];
  return a;
}

// B-fragment: base points at (n=0, k=0) of an n-major [16][k] region, stride ldk
__device__ __forceinline__ v16bf frag_b(const __bf16* base, int ldk, int lane) {
  const int n = lane & 15, half = lane >> 4;
  const __bf16* p = base + (size_t)n * ldk + 8 * half;
  v16bf b;
#pragma unroll
  for (int e = 0; e < 8; ++e) b[e] = p[e];
#pragma unroll
  for (int e = 0; e < 8; ++e) b[8 + e] = p[16 + e];
  return b;
}

// ---------------------------------------------------------------------------
// Kernel 1/5: fp32-in, bf16-compute GEMM   C[M,N] = A[M,K] * W[K,N]
// 256 threads = 8 waves; macro tile 128x128; wave tile 32(M) x 64(N); K-step 32
// Double-buffered LDS: issue next tile's global loads, compute WMMAs from the
// current buffer while they are in flight, then cvt+store to the other buffer.
// ---------------------------------------------------------------------------
__global__ __launch_bounds__(256) void gemm_f32_bf16(
    const float* __restrict__ A, const float* __restrict__ W,
    float* __restrict__ C, int M, int N, int K) {
  __shared__ __bf16 As[2][128][32];   // [buf][m][k]
  __shared__ __bf16 Bs[2][128][32];   // [buf][n][k]  (W transposed into LDS)

  const int n0 = blockIdx.x * 128;
  const int m0 = blockIdx.y * 128;
  const int tid = threadIdx.x, lane = tid & 31, wid = tid >> 5;
  const int wm = wid & 3;          // 4 waves along M
  const int wn = wid >> 2;         // 2 waves along N

  // staging assignments (fixed per thread)
  const int amm = tid >> 1, akk = (tid & 1) * 16;   // A: 2 thr/row, 16 k each
  const int bnn = tid & 127, bkk = (tid >> 7) * 16; // W: thread owns column nn

  auto stage_global = [&](int k0, float* ra, float* rb) {
    const float* srcA = A + (size_t)(m0 + amm) * K + k0 + akk;
#pragma unroll
    for (int i = 0; i < 16; ++i) ra[i] = srcA[i];
#pragma unroll
    for (int i = 0; i < 16; ++i)
      rb[i] = W[(size_t)(k0 + bkk + i) * N + n0 + bnn];
  };
  auto stage_lds = [&](int buf, const float* ra, const float* rb) {
#pragma unroll
    for (int i = 0; i < 16; ++i) As[buf][amm][akk + i] = (__bf16)ra[i];
#pragma unroll
    for (int i = 0; i < 16; ++i) Bs[buf][bnn][bkk + i] = (__bf16)rb[i];
  };

  v8f acc[2][4];
  const v8f vzero = {};
#pragma unroll
  for (int i = 0; i < 2; ++i)
#pragma unroll
    for (int j = 0; j < 4; ++j) acc[i][j] = vzero;

  // prologue: stage tile 0
  {
    float ra[16], rb[16];
    stage_global(0, ra, rb);
    stage_lds(0, ra, rb);
  }
  __syncthreads();

  for (int k0 = 0; k0 < K; k0 += 32) {
    const int cur = (k0 >> 5) & 1;
    const bool more = (k0 + 32) < K;

    // issue next tile's global loads first (in flight during WMMAs)
    float na[16], nb[16];
    if (more) stage_global(k0 + 32, na, nb);
    if (k0 + 64 < K) {  // pull tile k0+64 into GL2 early (global_prefetch_b8)
      __builtin_prefetch(A + (size_t)(m0 + amm) * K + k0 + 64 + akk, 0, 0);
      __builtin_prefetch(W + (size_t)(k0 + 64 + bkk) * N + n0 + bnn, 0, 0);
    }

    // compute from current buffer
    v16bf af[2], bf_[4];
#pragma unroll
    for (int i = 0; i < 2; ++i)
      af[i] = frag_a(&As[cur][wm * 32 + i * 16][0], 32, lane);
#pragma unroll
    for (int j = 0; j < 4; ++j)
      bf_[j] = frag_b(&Bs[cur][wn * 64 + j * 16][0], 32, lane);
#pragma unroll
    for (int i = 0; i < 2; ++i)
#pragma unroll
      for (int j = 0; j < 4; ++j) acc[i][j] = wmma_bf16(af[i], bf_[j], acc[i][j]);

    // convert and store next tile into the other buffer, then one barrier
    if (more) stage_lds(cur ^ 1, na, nb);
    __syncthreads();
  }

  const int n = lane & 15, half = lane >> 4;
#pragma unroll
  for (int i = 0; i < 2; ++i)
#pragma unroll
    for (int j = 0; j < 4; ++j)
#pragma unroll
      for (int r = 0; r < 8; ++r) {
        const int m = m0 + wm * 32 + i * 16 + r + 8 * half;
        const int c = n0 + wn * 64 + j * 16 + n;
        C[(size_t)m * N + c] = acc[i][j][r];
      }
}

// ---------------------------------------------------------------------------
// Kernel 2/5: RoPE + pack Q[b][h][s][d], K[b][kv][s][d] as bf16
// one thread per (b, s, head, freq-pair j)
// ---------------------------------------------------------------------------
__global__ __launch_bounds__(256) void rope_pack(
    const float* __restrict__ qkv, __bf16* __restrict__ Qb,
    __bf16* __restrict__ Kb) {
  const int idx = blockIdx.x * 256 + threadIdx.x;
  const int j  = idx & 63;
  const int hh = (idx >> 6) % (NH + NKV);
  const int s  = ((idx >> 6) / (NH + NKV)) % SEQ;
  const int b  = idx / (64 * (NH + NKV) * SEQ);

  // inv_freq[j] = theta^(-j/64) = 2^(-j/64 * log2(1e6))
  const float invf = exp2f(-(float)j * (19.931568569324174f / 64.0f));
  const float ang = (float)s * invf;
  float sn, cs;
  sincosf(ang, &sn, &cs);

  const float* base;
  __bf16* dst;
  if (hh < NH) {
    base = qkv + (size_t)(b * SEQ + s) * QKVN + hh * HD;
    dst  = Qb + ((size_t)(b * NH + hh) * SEQ + s) * HD;
  } else {
    const int kv = hh - NH;
    base = qkv + (size_t)(b * SEQ + s) * QKVN + NH * HD + kv * HD;
    dst  = Kb + ((size_t)(b * NKV + kv) * SEQ + s) * HD;
  }
  const float x0 = base[j], x1 = base[j + 64];
  dst[j]      = (__bf16)(x0 * cs - x1 * sn);   // d < 64: x*cos - x[d+64]*sin
  dst[j + 64] = (__bf16)(x1 * cs + x0 * sn);   // d >=64: x*cos + x[d-64]*sin
}

// ---------------------------------------------------------------------------
// Kernel 3/5: pack V transposed: Vt[b][kv][d][s] bf16 (contiguous keys per d)
// ---------------------------------------------------------------------------
__global__ __launch_bounds__(256) void v_pack(
    const float* __restrict__ qkv, __bf16* __restrict__ Vt) {
  const int idx = blockIdx.x * 256 + threadIdx.x;
  const int s  = idx & (SEQ - 1);
  const int d  = (idx >> 11) & (HD - 1);
  const int kv = (idx >> 18) & (NKV - 1);
  const int b  = idx >> 20;
  const float v = qkv[(size_t)(b * SEQ + s) * QKVN + (NH + NKV) * HD + kv * HD + d];
  Vt[((size_t)(b * NKV + kv) * HD + d) * SEQ + s] = (__bf16)v;
}

// ---------------------------------------------------------------------------
// Kernel 4/5: flash attention, sliding-window causal (WIN=512), GQA 4:1
// 1 workgroup = (b, h, 64-query block); 4 waves x 16 query rows each.
// K/V staged in LDS per 32-key block; scores & PV via bf16 WMMA.
// ---------------------------------------------------------------------------
__global__ __launch_bounds__(128) void attn_kernel(
    const __bf16* __restrict__ Qb, const __bf16* __restrict__ Kb,
    const __bf16* __restrict__ Vt, float* __restrict__ attn) {
  const int nqb = SEQ / 64;                       // 32
  const int blk = blockIdx.x;
  const int qb  = blk % nqb;
  const int h   = (blk / nqb) % NH;
  const int b   = blk / (nqb * NH);
  const int kvh = h >> 2;                         // n_rep = 4

  const int lane = threadIdx.x & 31, wid = threadIdx.x >> 5;
  const int q0 = qb * 64 + wid * 16;

  __shared__ __bf16 kt[32][128];                  // [key][d]
  __shared__ __bf16 vt[128][32];                  // [d][key]  (V^T)
  __shared__ __bf16 plds[4][16][32];              // per-wave P scratch

  // Q fragments for this wave's 16 rows, 4 chunks of K-dim 32
  const __bf16* qbase = Qb + ((size_t)(b * NH + h) * SEQ + q0) * HD;
  v16bf qf[4];
#pragma unroll
  for (int c = 0; c < 4; ++c) qf[c] = frag_a(qbase + 32 * c, HD, lane);

  const v8f vzero = {};
  v8f o[8];
#pragma unroll
  for (int t = 0; t < 8; ++t) o[t] = vzero;
  float mi[8], li[8];
#pragma unroll
  for (int r = 0; r < 8; ++r) { mi[r] = -1e30f; li[r] = 0.0f; }

  const __bf16* kbase = Kb + ((size_t)(b * NKV + kvh) * SEQ) * HD;
  const __bf16* vbase = Vt + ((size_t)(b * NKV + kvh) * HD) * SEQ;

  const int qblk0 = qb * 64;
  const int ksr = qblk0 - (WIN - 1);
  const int ks = (ksr < 0) ? 0 : (ksr & ~31);
  const int n = lane & 15, half = lane >> 4;

  for (int kb = ks; kb < qblk0 + 64; kb += 32) {
    __syncthreads();
    {  // stage K block: 32 keys x 128 d (4 threads per key, 32 d each)
      const int key = threadIdx.x >> 2, d0 = (threadIdx.x & 3) * 32;
      const uint4* s4 = (const uint4*)(kbase + (size_t)(kb + key) * HD + d0);
      uint4* dst = (uint4*)&kt[key][d0];
#pragma unroll
      for (int i = 0; i < 4; ++i) dst[i] = s4[i];
      if (kb + 32 < qblk0 + 64)   // prefetch next K block into GL2
        __builtin_prefetch(kbase + (size_t)(kb + 32 + key) * HD + d0, 0, 0);
    }
    {  // stage V^T block: 128 d rows x 32 keys (1 thread per d row)
      const int d = threadIdx.x;
      const uint4* s4 = (const uint4*)(vbase + (size_t)d * SEQ + kb);
      uint4* dst = (uint4*)&vt[d][0];
#pragma unroll
      for (int i = 0; i < 4; ++i) dst[i] = s4[i];
      if (kb + 32 < qblk0 + 64)   // prefetch next V block into GL2
        __builtin_prefetch(vbase + (size_t)d * SEQ + kb + 32, 0, 0);
    }
    __syncthreads();

    // scores: S = Q(16x128) . K^T -> two 16x16 tiles (keys kb..+15, kb+16..+31)
    v8f sA = vzero, sB = vzero;
#pragma unroll
    for (int c = 0; c < 4; ++c) {
      v16bf bA = frag_b(&kt[0][32 * c], HD, lane);
      v16bf bB = frag_b(&kt[16][32 * c], HD, lane);
      sA = wmma_bf16(qf[c], bA, sA);
      sB = wmma_bf16(qf[c], bB, sB);
    }

    // online softmax update + P staging (C-layout -> A-layout via LDS)
#pragma unroll
    for (int r = 0; r < 8; ++r) {
      const int qrow = q0 + r + 8 * half;
      const int kA = kb + n, kB = kb + 16 + n;
      float a = sA[r] * SCALE, bb = sB[r] * SCALE;
      a  = ((kA <= qrow) && (qrow - kA < WIN)) ? a : -1e30f;
      bb = ((kB <= qrow) && (qrow - kB < WIN)) ? bb : -1e30f;
      float mx = fmaxf(a, bb);
#pragma unroll
      for (int mk = 1; mk < 16; mk <<= 1) mx = fmaxf(mx, __shfl_xor(mx, mk, 32));
      const float newm = fmaxf(mi[r], mx);
      const float cr = __expf(mi[r] - newm);
      const float pa = __expf(a - newm), pb = __expf(bb - newm);
      float ls = pa + pb;
#pragma unroll
      for (int mk = 1; mk < 16; mk <<= 1) ls += __shfl_xor(ls, mk, 32);
      li[r] = li[r] * cr + ls;
      mi[r] = newm;
      plds[wid][r + 8 * half][n]      = (__bf16)pa;
      plds[wid][r + 8 * half][n + 16] = (__bf16)pb;
#pragma unroll
      for (int t = 0; t < 8; ++t) o[t][r] *= cr;
    }

    // O += P(16x32) . V(32x128)
    v16bf pf = frag_a(&plds[wid][0][0], 32, lane);
#pragma unroll
    for (int t = 0; t < 8; ++t) {
      v16bf bv = frag_b(&vt[t * 16][0], 32, lane);
      o[t] = wmma_bf16(pf, bv, o[t]);
    }
  }

  // epilogue: normalize and write attn[b][s][h*128 + d] (fp32, GEMM2 input)
#pragma unroll
  for (int r = 0; r < 8; ++r) {
    const float inv = 1.0f / li[r];
    const size_t row = (size_t)(b * SEQ + q0 + r + 8 * half);
#pragma unroll
    for (int t = 0; t < 8; ++t)
      attn[row * (NH * HD) + h * HD + t * 16 + n] = o[t][r] * inv;
  }
}

// ---------------------------------------------------------------------------
// Host launch
// ---------------------------------------------------------------------------
extern "C" void kernel_launch(void* const* d_in, const int* in_sizes, int n_in,
                              void* d_out, int out_size, void* d_ws, size_t ws_size,
                              hipStream_t stream) {
  const float* x     = (const float*)d_in[0];
  const float* w_qkv = (const float*)d_in[1];
  const float* w_o   = (const float*)d_in[2];
  float* out = (float*)d_out;

  char* ws = (char*)d_ws;
  size_t off = 0;
  float* qkv = (float*)(ws + off);  off += (size_t)BATCH * SEQ * QKVN * 4;          // 50.3 MB
  __bf16* Qb = (__bf16*)(ws + off); off += (size_t)BATCH * NH * SEQ * HD * 2;       // 16.8 MB
  __bf16* Kb = (__bf16*)(ws + off); off += (size_t)BATCH * NKV * SEQ * HD * 2;      //  4.2 MB
  __bf16* Vt = (__bf16*)(ws + off); off += (size_t)BATCH * NKV * SEQ * HD * 2;      //  4.2 MB
  float* attn = (float*)(ws + off);                                                 // 33.5 MB

  const int M = BATCH * SEQ;  // 4096

  // 1) QKV projection: [4096,2048] x [2048,3072]
  gemm_f32_bf16<<<dim3(QKVN / 128, M / 128), 256, 0, stream>>>(
      x, w_qkv, qkv, M, QKVN, EMB);

  // 2) RoPE + pack Q/K (B*S*(H+KV)*64 threads)
  rope_pack<<<(BATCH * SEQ * (NH + NKV) * 64) / 256, 256, 0, stream>>>(qkv, Qb, Kb);

  // 3) pack V^T (B*KV*D*S threads)
  v_pack<<<(BATCH * NKV * HD * SEQ) / 256, 256, 0, stream>>>(qkv, Vt);

  // 4) windowed flash attention: 1 WG per (b,h,64-row block)
  attn_kernel<<<BATCH * NH * (SEQ / 64), 128, 0, stream>>>(Qb, Kb, Vt, attn);

  // 5) output projection: [4096,2048] x [2048,2048]
  gemm_f32_bf16<<<dim3(EMB / 128, M / 128), 256, 0, stream>>>(
      attn, w_o, out, M, EMB, EMB);
}
